// SupportPairMapper_13022340842028
// MI455X (gfx1250) — matrix operand order repaired and checked by reading, not compiled, not run on hardware
//
#include <hip/hip_runtime.h>
#include <hip/hip_bf16.h>

// ---------------------------------------------------------------------------
// Types
// ---------------------------------------------------------------------------
typedef __bf16 bf16;
typedef bf16  bf16x16 __attribute__((ext_vector_type(16)));
typedef bf16  bf16x8  __attribute__((ext_vector_type(8)));
typedef float floatx8 __attribute__((ext_vector_type(8)));
typedef unsigned int u32;
typedef u32 u32x4 __attribute__((ext_vector_type(4)));
typedef int i32x4 __attribute__((ext_vector_type(4)));
typedef int i32x8 __attribute__((ext_vector_type(8)));

#define LL 2048
#define HH 2048
#define NHEADS 16
#define HD 128

static __device__ __forceinline__ floatx8 wmma_bf16(const bf16x16& a,
                                                    const bf16x16& b,
                                                    const floatx8& c) {
  return __builtin_amdgcn_wmma_f32_16x16x32_bf16(
      /*neg_a=*/false, a, /*neg_b=*/false, b,
      /*c_mod=*/(short)0, c, /*reuse_a=*/false, /*reuse_b=*/false);
}

// Assemble the 16x32 bf16 A-fragment from two 8-element chunks:
// lane<16: row=lid, K 0..15 ; lane>=16: row=lid, K 16..31
static __device__ __forceinline__ bf16x16 load_a_frag(const bf16* rowptr,
                                                      int half) {
  bf16x8 lo = *(const bf16x8*)(rowptr + half * 16);
  bf16x8 hi = *(const bf16x8*)(rowptr + half * 16 + 8);
  bf16x16 a;
#pragma unroll
  for (int i = 0; i < 8; ++i) { a[i] = lo[i]; a[i + 8] = hi[i]; }
  return a;
}

// ---------------------------------------------------------------------------
// Tensor Data Mover: 2-D tile load Global -> LDS (clang-23 6-arg builtin:
// (u32x4 g0, i32x8 g1, i32x4, i32x4, i32x8, i32 cpol)).
// D# per CDNA5 ISA:
//   group0: [1:0]=count=1 | [63:32]=lds_addr | [120:64]=global_addr | type=2
//   group1: data_size(2B) @ [17:16], tensor_dim0 @ [79:48], tensor_dim1 @
//           [111:80], tile_dim0 @ [127:112], tile_dim1 @ [143:128],
//           tensor_dim0_stride @ [207:160]
//   remaining groups: zero (2-D tile; dims >=2 unused)
// Tile is stored compacted row-major in LDS starting at lds_byte_addr.
// ---------------------------------------------------------------------------
static __device__ __forceinline__ void tdm_load_2d(const bf16* gsrc,
                                                   u32 lds_byte_addr,
                                                   u32 tile_d0_elems,   // 32
                                                   u32 tile_d1_rows,    // 128
                                                   u32 tensor_d0,       // K
                                                   u32 tensor_d1,       // rows
                                                   u32 stride0_elems) { // K
  unsigned long long ga = (unsigned long long)gsrc;
  u32x4 g0;
  g0[0] = 1u;                                            // count=1 (user D#)
  g0[1] = lds_byte_addr;                                 // lds_addr
  g0[2] = (u32)(ga & 0xffffffffu);                       // global_addr[31:0]
  g0[3] = (u32)((ga >> 32) & 0x01ffffffu) | (2u << 30);  // addr[56:32]|type=2
  i32x8 g1;
  g1[0] = (int)(1u << 16);                               // data_size=1 -> 2 B
  g1[1] = (int)((tensor_d0 & 0xffffu) << 16);            // td0[15:0]
  g1[2] = (int)((tensor_d0 >> 16) | ((tensor_d1 & 0xffffu) << 16));
  g1[3] = (int)((tensor_d1 >> 16) | (tile_d0_elems << 16));
  g1[4] = (int)(tile_d1_rows & 0xffffu);                 // tile_dim1, tile_dim2=0
  g1[5] = (int)stride0_elems;                            // stride0[31:0]
  g1[6] = 0;                                             // stride0[47:32]=0
  g1[7] = 0;
  i32x4 z4;
  z4[0] = 0; z4[1] = 0; z4[2] = 0; z4[3] = 0;
  i32x8 z8;
#pragma unroll
  for (int i = 0; i < 8; ++i) z8[i] = 0;
  __builtin_amdgcn_tensor_load_to_lds(g0, g1, z4, z4, z8, 0);
}

// ---------------------------------------------------------------------------
// Elementwise clamp to +-100 (reference guard), 8 bf16 per thread (b128)
// ---------------------------------------------------------------------------
__global__ void __launch_bounds__(256)
clamp_kernel(const bf16* __restrict__ in, bf16* __restrict__ out, int n8) {
  int i = blockIdx.x * 256 + threadIdx.x;
  if (i < n8) {
    bf16x8 v = *(const bf16x8*)(in + (size_t)i * 8);
    bf16x8 r;
#pragma unroll
    for (int j = 0; j < 8; ++j) {
      float f = (float)v[j];
      r[j] = (bf16)fminf(fmaxf(f, -100.0f), 100.0f);
    }
    *(bf16x8*)(out + (size_t)i * 8) = r;
  }
}

// ---------------------------------------------------------------------------
// C[M,N] = A[M,K] @ B[N,K]^T   (torch Linear: x @ W.T)
// Block = 4 waves computing a 128x128 tile (2x2 grid of 64x64 wave tiles,
// each a 4x4 grid of 16x16 WMMA tiles -> 64 v_wmma per block per k-step).
// A/B 128x32 tiles are staged into double-buffered LDS by the Tensor Data
// Mover (tensor_load_to_lds, TENSORcnt), overlapping DMA with WMMA; waves
// read fragments from LDS via ds_load_b128.
// trans_out=1 stores C transposed (C_t[N,M]) -- used to produce V^T.
// ---------------------------------------------------------------------------
__global__ void __launch_bounds__(128)
gemm_nt_bf16(const bf16* __restrict__ A, const bf16* __restrict__ B,
             bf16* __restrict__ C, int M, int N, int K, int trans_out) {
  __shared__ __align__(32) bf16 Abuf[2][128 * 32];  // 2 x 8 KB
  __shared__ __align__(32) bf16 Bbuf[2][128 * 32];  // 2 x 8 KB

  const int tid  = threadIdx.x;
  const int lane = tid & 31;
  const int wave = tid >> 5;
  const int wm   = wave >> 1;       // wave row (0..1)
  const int wn   = wave & 1;        // wave col (0..1)
  const int half = lane >> 4;
  const int lid  = lane & 15;
  const int n0 = blockIdx.x * 128;
  const int m0 = blockIdx.y * 128;

  const u32 aoff[2] = { (u32)(size_t)&Abuf[0][0], (u32)(size_t)&Abuf[1][0] };
  const u32 boff[2] = { (u32)(size_t)&Bbuf[0][0], (u32)(size_t)&Bbuf[1][0] };

  floatx8 acc[4][4];
#pragma unroll
  for (int mt = 0; mt < 4; ++mt)
#pragma unroll
    for (int nt = 0; nt < 4; ++nt)
#pragma unroll
      for (int r = 0; r < 8; ++r) acc[mt][nt][r] = 0.0f;

  // Preload k=0 tiles into buffer 0 (wave 0 drives the TDM).
  if (wave == 0) {
    tdm_load_2d(A + (size_t)m0 * K, aoff[0], 32, 128, (u32)K, (u32)M, (u32)K);
    tdm_load_2d(B + (size_t)n0 * K, boff[0], 32, 128, (u32)K, (u32)N, (u32)K);
  }
  __builtin_amdgcn_s_wait_tensorcnt(0);
  __syncthreads();

  int buf = 0;
  for (int k = 0; k < K; k += 32) {
    const int nb = buf ^ 1;
    if (k + 32 < K && wave == 0) {
      tdm_load_2d(A + (size_t)m0 * K + (k + 32), aoff[nb], 32, 128, (u32)K,
                  (u32)M, (u32)K);
      tdm_load_2d(B + (size_t)n0 * K + (k + 32), boff[nb], 32, 128, (u32)K,
                  (u32)N, (u32)K);
    }

    // Fragments from LDS (tile rows are 32 bf16 = 64 B pitch).
    bf16x16 a[4], b[4];
#pragma unroll
    for (int t = 0; t < 4; ++t) {
      const bf16* ap = &Abuf[buf][(wm * 64 + t * 16 + lid) * 32];
      a[t] = load_a_frag(ap, half);
      const bf16* bp = &Bbuf[buf][(wn * 64 + t * 16 + lid) * 32 + half * 16];
      b[t] = *(const bf16x16*)bp;
    }
#pragma unroll
    for (int mt = 0; mt < 4; ++mt)
#pragma unroll
      for (int nt = 0; nt < 4; ++nt)
        acc[mt][nt] = wmma_bf16(a[mt], b[nt], acc[mt][nt]);

    __builtin_amdgcn_s_wait_tensorcnt(0);  // issuing wave: DMA done
    __syncthreads();                       // all waves: done reading `buf`
    buf = nb;
  }

  const int mbase = m0 + wm * 64;
  const int nbase = n0 + wn * 64;
  if (!trans_out) {
#pragma unroll
    for (int mt = 0; mt < 4; ++mt)
#pragma unroll
      for (int nt = 0; nt < 4; ++nt)
#pragma unroll
        for (int r = 0; r < 8; ++r)
          C[(size_t)(mbase + mt * 16 + half * 8 + r) * N + nbase + nt * 16 + lid] =
              (bf16)acc[mt][nt][r];
  } else {
#pragma unroll
    for (int mt = 0; mt < 4; ++mt)
#pragma unroll
      for (int nt = 0; nt < 4; ++nt)
#pragma unroll
        for (int r = 0; r < 8; ++r)
          C[(size_t)(nbase + nt * 16 + lid) * M + mbase + mt * 16 + half * 8 + r] =
              (bf16)acc[mt][nt][r];
  }
}

// ---------------------------------------------------------------------------
// Flash-style cross attention per head.
// Grid: (L/16, NHEADS), block = 1 wave.
// Q[L,H], Km[L,H] row-major; Vt[H,L] (V transposed) so the A*V pass reads
// contiguous B fragments. Output Z[L,H] bf16.
// ---------------------------------------------------------------------------
__global__ void __launch_bounds__(32)
attn_kernel(const bf16* __restrict__ Q, const bf16* __restrict__ Km,
            const bf16* __restrict__ Vt, bf16* __restrict__ Z) {
  __shared__ bf16 Pstage[16 * 32];   // P tile staged to convert D-layout -> A-layout

  const int lane = threadIdx.x;
  const int half = lane >> 4;
  const int lid  = lane & 15;
  const int h  = blockIdx.y;
  const int l0 = blockIdx.x * 16;
  const float scale = 0.08838834764831843f;  // 1/sqrt(HEAD_DIM)

  // Preload Q tile (16 x 128) as 4 A-fragments (K steps of 32)
  bf16x16 qa[4];
  const bf16* qrow = Q + (size_t)(l0 + lid) * HH + h * HD;
#pragma unroll
  for (int kk = 0; kk < 4; ++kk) qa[kk] = load_a_frag(qrow + kk * 32, half);

  // Output accumulator: 16 rows x 128 cols fp32 = 8 n-tiles of 16x16
  floatx8 O[8];
#pragma unroll
  for (int t = 0; t < 8; ++t)
#pragma unroll
    for (int r = 0; r < 8; ++r) O[t][r] = 0.0f;

  float rowmax[8], rowsum[8];
#pragma unroll
  for (int r = 0; r < 8; ++r) { rowmax[r] = -1e30f; rowsum[r] = 0.0f; }

  for (int m0 = 0; m0 < LL; m0 += 32) {
    const bf16* k0row = Km + (size_t)(m0 + lid) * HH + h * HD;
    const bf16* k1row = Km + (size_t)(m0 + 16 + lid) * HH + h * HD;
    // gfx1250 prefetch path (global_prefetch_b8) for the next m-block
    __builtin_prefetch(k0row + (size_t)32 * HH, 0, 0);
    __builtin_prefetch(k1row + (size_t)32 * HH, 0, 0);

    // ---- S tile = Q (16x128) . K-block(32x128)^T, two 16x16 accumulators ----
    floatx8 s0, s1;
#pragma unroll
    for (int r = 0; r < 8; ++r) { s0[r] = 0.0f; s1[r] = 0.0f; }
#pragma unroll
    for (int kk = 0; kk < 4; ++kk) {
      bf16x16 b0 = *(const bf16x16*)(k0row + kk * 32 + half * 16);
      bf16x16 b1 = *(const bf16x16*)(k1row + kk * 32 + half * 16);
      s0 = wmma_bf16(qa[kk], b0, s0);
      s1 = wmma_bf16(qa[kk], b1, s1);
    }

    // ---- scale, clamp(+-50), online softmax ----
    float tmax[8];
#pragma unroll
    for (int r = 0; r < 8; ++r) {
      float a = fminf(fmaxf(s0[r] * scale, -50.0f), 50.0f);
      float b = fminf(fmaxf(s1[r] * scale, -50.0f), 50.0f);
      s0[r] = a; s1[r] = b;
      tmax[r] = fmaxf(a, b);
    }
    // reduce over the 16 lanes holding one row (xor masks 1,2,4,8 stay in-group)
#pragma unroll
    for (int off = 1; off < 16; off <<= 1)
#pragma unroll
      for (int r = 0; r < 8; ++r) tmax[r] = fmaxf(tmax[r], __shfl_xor(tmax[r], off, 32));

    float corr[8];
#pragma unroll
    for (int r = 0; r < 8; ++r) {
      float nm = fmaxf(rowmax[r], tmax[r]);
      corr[r] = __expf(rowmax[r] - nm);
      rowmax[r] = nm;
    }

    float psum[8];
#pragma unroll
    for (int r = 0; r < 8; ++r) {
      float p0 = __expf(s0[r] - rowmax[r]);
      float p1 = __expf(s1[r] - rowmax[r]);
      psum[r] = p0 + p1;
      Pstage[(half * 8 + r) * 32 + lid]      = (bf16)p0;
      Pstage[(half * 8 + r) * 32 + 16 + lid] = (bf16)p1;
    }
#pragma unroll
    for (int off = 1; off < 16; off <<= 1)
#pragma unroll
      for (int r = 0; r < 8; ++r) psum[r] += __shfl_xor(psum[r], off, 32);
#pragma unroll
    for (int r = 0; r < 8; ++r) rowsum[r] = rowsum[r] * corr[r] + psum[r];

    __syncthreads();
    // Re-read P in A-fragment layout (lane = row, elements = m-slice)
    bf16x16 pa;
    {
      const bf16* prow = &Pstage[lid * 32 + half * 16];
#pragma unroll
      for (int i = 0; i < 8; ++i) { pa[i] = prow[i]; pa[i + 8] = prow[i + 8]; }
    }

    // ---- O = O*corr + P(16x32) . V-block(32x128) ----
#pragma unroll
    for (int t = 0; t < 8; ++t) {
#pragma unroll
      for (int r = 0; r < 8; ++r) O[t][r] *= corr[r];
      const bf16* vtr = Vt + (size_t)(h * HD + t * 16 + lid) * LL + m0 + half * 16;
      bf16x16 vb = *(const bf16x16*)vtr;
      O[t] = wmma_bf16(pa, vb, O[t]);
    }
    __syncthreads();  // protect Pstage before next iteration
  }

  // ---- normalize rows and store bf16 ----
#pragma unroll
  for (int t = 0; t < 8; ++t)
#pragma unroll
    for (int r = 0; r < 8; ++r) {
      float v = O[t][r] / rowsum[r];
      Z[(size_t)(l0 + half * 8 + r) * HH + h * HD + t * 16 + lid] = (bf16)v;
    }
}

// ---------------------------------------------------------------------------
// Latent-query pooling softmax + LayerNorm. Single block of 256 threads;
// total work here is ~17 MFLOP (negligible).
// ---------------------------------------------------------------------------
__global__ void __launch_bounds__(256)
pool_ln_kernel(const bf16* __restrict__ Zo, const bf16* __restrict__ lq,
               const bf16* __restrict__ gamma, const bf16* __restrict__ beta,
               bf16* __restrict__ out) {
  __shared__ float red[256];
  __shared__ float attn_sh[LL];
  const int t = threadIdx.x;
  const float iscale = 0.022097086912079608f;  // 1/sqrt(H)

  // s[l] = Zo[l,:] . lq / sqrt(H), clamp +-50
  float sv[8];
  float smax_l = -1e30f;
#pragma unroll
  for (int i = 0; i < 8; ++i) {
    int l = t * 8 + i;
    const bf16* zr = Zo + (size_t)l * HH;
    float acc = 0.0f;
    for (int d = 0; d < HH; ++d) acc += (float)zr[d] * (float)lq[d];
    acc = fminf(fmaxf(acc * iscale, -50.0f), 50.0f);
    sv[i] = acc;
    smax_l = fmaxf(smax_l, acc);
  }
  red[t] = smax_l; __syncthreads();
  for (int off = 128; off > 0; off >>= 1) {
    if (t < off) red[t] = fmaxf(red[t], red[t + off]);
    __syncthreads();
  }
  float smax = red[0]; __syncthreads();

  float lsum = 0.0f;
#pragma unroll
  for (int i = 0; i < 8; ++i) {
    float e = __expf(sv[i] - smax);
    attn_sh[t * 8 + i] = e;
    lsum += e;
  }
  red[t] = lsum; __syncthreads();
  for (int off = 128; off > 0; off >>= 1) {
    if (t < off) red[t] += red[t + off];
    __syncthreads();
  }
  float denom = red[0]; __syncthreads();

  // m[d] = sum_l attn[l] * Zo[l,d]  (each thread owns 8 columns)
  float mv[8];
#pragma unroll
  for (int i = 0; i < 8; ++i) mv[i] = 0.0f;
  for (int l = 0; l < LL; ++l) {
    float a = attn_sh[l] / denom;
    const bf16* zr = Zo + (size_t)l * HH + t * 8;
#pragma unroll
    for (int i = 0; i < 8; ++i) mv[i] += a * (float)zr[i];
  }
#pragma unroll
  for (int i = 0; i < 8; ++i) mv[i] = (float)(bf16)mv[i];  // ref rounds m to bf16

  // LayerNorm over H (fp32 stats)
  float psum = 0.0f;
#pragma unroll
  for (int i = 0; i < 8; ++i) psum += mv[i];
  red[t] = psum; __syncthreads();
  for (int off = 128; off > 0; off >>= 1) {
    if (t < off) red[t] += red[t + off];
    __syncthreads();
  }
  float mu = red[0] / (float)HH; __syncthreads();

  float vsum = 0.0f;
#pragma unroll
  for (int i = 0; i < 8; ++i) { float d = mv[i] - mu; vsum += d * d; }
  red[t] = vsum; __syncthreads();
  for (int off = 128; off > 0; off >>= 1) {
    if (t < off) red[t] += red[t + off];
    __syncthreads();
  }
  float var = red[0] / (float)HH;
  float rinv = rsqrtf(var + 1e-5f);

#pragma unroll
  for (int i = 0; i < 8; ++i) {
    int d = t * 8 + i;
    float nm = (mv[i] - mu) * rinv;
    out[d] = (bf16)((float)(bf16)nm * (float)gamma[d] + (float)beta[d]);
  }
}

// ---------------------------------------------------------------------------
// Launch
// ---------------------------------------------------------------------------
extern "C" void kernel_launch(void* const* d_in, const int* in_sizes, int n_in,
                              void* d_out, int out_size, void* d_ws, size_t ws_size,
                              hipStream_t stream) {
  const bf16* x     = (const bf16*)d_in[0];
  const bf16* y     = (const bf16*)d_in[1];
  const bf16* Wq    = (const bf16*)d_in[2];
  const bf16* Wk    = (const bf16*)d_in[3];
  const bf16* Wv    = (const bf16*)d_in[4];
  const bf16* Wo    = (const bf16*)d_in[5];
  const bf16* lq    = (const bf16*)d_in[6];
  const bf16* gamma = (const bf16*)d_in[7];
  const bf16* beta  = (const bf16*)d_in[8];
  bf16* out = (bf16*)d_out;

  char* ws = (char*)d_ws;
  const size_t mat = (size_t)LL * HH * sizeof(bf16);  // 8 MB
  bf16* xc = (bf16*)(ws + 0 * mat);
  bf16* yc = (bf16*)(ws + 1 * mat);
  bf16* Qm = (bf16*)(ws + 2 * mat);
  bf16* Km = (bf16*)(ws + 3 * mat);
  bf16* Vt = (bf16*)(ws + 4 * mat);   // V transposed: [H, L]
  bf16* Zm = (bf16*)(ws + 5 * mat);
  bf16* Zo = (bf16*)(ws + 6 * mat);

  const int n8 = (LL * HH) / 8;
  clamp_kernel<<<n8 / 256, 256, 0, stream>>>(x, xc, n8);
  clamp_kernel<<<n8 / 256, 256, 0, stream>>>(y, yc, n8);

  dim3 gg(HH / 128, LL / 128);
  gemm_nt_bf16<<<gg, 128, 0, stream>>>(xc, Wq, Qm, LL, HH, HH, 0);
  gemm_nt_bf16<<<gg, 128, 0, stream>>>(yc, Wk, Km, LL, HH, HH, 0);
  gemm_nt_bf16<<<gg, 128, 0, stream>>>(yc, Wv, Vt, LL, HH, HH, 1);

  attn_kernel<<<dim3(LL / 16, NHEADS), 32, 0, stream>>>(Qm, Km, Vt, Zm);

  gemm_nt_bf16<<<gg, 128, 0, stream>>>(Zm, Wo, Zo, LL, HH, HH, 0);

  pool_ln_kernel<<<1, 256, 0, stream>>>(Zo, lq, gamma, beta, out);
}
